// MultiHeadAttention_21973052686770
// MI455X (gfx1250) — compile-verified, matrix operand order checked by reading, not run on hardware
//
#include <hip/hip_runtime.h>
#include <hip/hip_bf16.h>

// ---------------------------------------------------------------------------
// MI455X (gfx1250) multi-head attention, bf16 WMMA pipeline.
// b=2, s=2048, d_model=2048, H=16, hd=128.
// Round 6: 6-arg tensor_load_to_lds (this toolchain's arity) for attention
// K/V staging with TENSORcnt double buffering; async-LDS staging in GEMM.
// ---------------------------------------------------------------------------

typedef __attribute__((ext_vector_type(16))) __bf16 v16bf;
typedef __attribute__((ext_vector_type(8)))  __bf16 v8bf;
typedef __attribute__((ext_vector_type(8)))  float  v8f;
typedef int      vint4 __attribute__((vector_size(16)));
typedef unsigned v4u   __attribute__((vector_size(16)));
typedef int      v8i_t __attribute__((vector_size(32)));
typedef int      v4i_t __attribute__((vector_size(16)));

#define BATCH   2
#define SEQ     2048
#define DMODEL  2048
#define NHEADS  16
#define HEADDIM 128
#define ROWS    (BATCH * SEQ)          // 4096 token rows

// ---- gfx1250 async-to-LDS availability (probe via __has_builtin) ----------
#if defined(__has_builtin)
#  if __has_builtin(__builtin_amdgcn_global_load_async_to_lds_b128) && \
      __has_builtin(__builtin_amdgcn_s_wait_asynccnt)
#    define HAVE_ASYNC_LDS 1
#  endif
#  if __has_builtin(__builtin_amdgcn_tensor_load_to_lds) && \
      __has_builtin(__builtin_amdgcn_s_wait_tensorcnt)
#    define HAVE_TDM 1
#  endif
#endif
#ifndef HAVE_ASYNC_LDS
#  define HAVE_ASYNC_LDS 0
#endif
#ifndef HAVE_TDM
#  define HAVE_TDM 0
#endif

#if HAVE_ASYNC_LDS
#  define ASYNC_CP16(dst, src)                                         \
    __builtin_amdgcn_global_load_async_to_lds_b128(                    \
        (__attribute__((address_space(1))) vint4*)(src),               \
        (__attribute__((address_space(3))) vint4*)(dst), 0, 0)
#  define ASYNC_WAIT(n) __builtin_amdgcn_s_wait_asynccnt(n)
#else
#  define ASYNC_CP16(dst, src) (*(v8bf*)(dst) = *(const v8bf*)(src))
#  define ASYNC_WAIT(n) ((void)0)
#endif

// ---- fragment helpers -----------------------------------------------------
// 16-bit A-frag (16x32, MxK): lane = row (lane&15); lane-half kh selects
// K subsets {kh*8..kh*8+7} and {16+kh*8..16+kh*8+7}.  B-frag (32x16, KxN)
// is the mirror with lane = column, so both A rows and B columns are
// contiguous 16B loads from row-major storage.
__device__ __forceinline__ v16bf pack16(v8bf lo, v8bf hi) {
    v16bf r;
#pragma unroll
    for (int j = 0; j < 8; ++j) { r[j] = lo[j]; r[8 + j] = hi[j]; }
    return r;
}

__device__ __forceinline__ v16bf load_frag(const __bf16* p, int kh) {
    v8bf lo = *(const v8bf*)(p + kh * 8);
    v8bf hi = *(const v8bf*)(p + 16 + kh * 8);
    return pack16(lo, hi);
}

__device__ __forceinline__ v8f wmma_bf16(v16bf a, v16bf b, v8f c) {
    return __builtin_amdgcn_wmma_f32_16x16x32_bf16(
        /*neg_a=*/false, a, /*neg_b=*/false, b,
        /*c_mod=*/(short)0, c, /*reuse_a=*/false, /*reuse_b=*/false);
}

#if HAVE_TDM
// byte offset of a shared-memory object within the LDS aperture
__device__ __forceinline__ unsigned lds_addr_of(const void* p) {
    return (unsigned)(unsigned long long)
        (__attribute__((address_space(3))) const void*)p;
}

// Issue one 2D TDM tile load: tile1 rows x tile0 elements (bf16), row pitch
// stride0 (elements), LDS padding padAmt+1 dwords every 2^(padInt+1) dwords.
// D# layout per CDNA5 ISA 8.3/8.4: group0 {count=1, lds_addr, global_addr,
// type=2}; group1 {data_size=1(2B), pad ctl, tensor dims, tile dims, stride}.
__device__ __forceinline__ void tdm_load_2d(const void* gaddr, unsigned ldsAddr,
                                            unsigned tdim0, unsigned tdim1,
                                            unsigned stride0,
                                            unsigned tile0, unsigned tile1,
                                            unsigned padInt, unsigned padAmt) {
    const unsigned long long ga = (unsigned long long)gaddr;
    v4u g0;
    g0[0] = 1u;                                        // count=1, user mode
    g0[1] = ldsAddr;                                   // lds_addr [63:32]
    g0[2] = (unsigned)(ga & 0xFFFFFFFFu);              // global_addr lo
    g0[3] = (unsigned)((ga >> 32) & 0x01FFFFFFu)       // global_addr hi (57b)
          | (2u << 30);                                // type=2 ("image")
    v8i_t g1;
    g1[0] = (int)((1u << 16)                           // data_size=2B
                | (1u << 20)                           // pad_enable
                | (padInt << 22) | (padAmt << 25));    // LDS padding
    g1[1] = (int)((tdim0 & 0xFFFFu) << 16);            // tensor_dim0 [79:48]
    g1[2] = (int)(((tdim0 >> 16) & 0xFFFFu)
                | ((tdim1 & 0xFFFFu) << 16));          // tensor_dim1 [111:80]
    g1[3] = (int)(((tdim1 >> 16) & 0xFFFFu)
                | ((tile0 & 0xFFFFu) << 16));          // tile_dim0 [127:112]
    g1[4] = (int)(tile1 & 0xFFFFu);                    // tile_dim1 [143:128]
    g1[5] = (int)stride0;                              // dim0_stride [207:160]
    g1[6] = 0;
    g1[7] = 0;
    v4i_t z4 = {0, 0, 0, 0};                           // 2D: groups 2/3 unused
    v8i_t z8 = {0, 0, 0, 0, 0, 0, 0, 0};               // extra group (unused)
    __builtin_amdgcn_tensor_load_to_lds(g0, g1, z4, z4, z8, 0);
}
#endif

// ---- fp32 -> bf16 cast ----------------------------------------------------
__global__ void cast_bf16_kernel(const float* __restrict__ in,
                                 __bf16* __restrict__ out, int n) {
    int i = blockIdx.x * blockDim.x + threadIdx.x;
    if (i < n) out[i] = (__bf16)in[i];
}

// ---- GEMM: C[M,N] = A[M,K] @ B[N,K]^T, bf16 in / fp32 out ----------------
// Block = 8 waves = 128x64 output tile.  B strip (64 rows of W) is staged
// through a double-buffered LDS tile in 64-wide K chunks via async copies;
// each wave computes a 16x64 strip, reading A directly from global (stream).
#define KC 64
#define LDSB_STRIDE 72   // 64 + 8 bf16 pad -> 16 distinct bank offsets

__global__ __launch_bounds__(256)
void gemm_bf16_nt_kernel(const __bf16* __restrict__ A,
                         const __bf16* __restrict__ B,
                         float* __restrict__ C, int M, int N, int K) {
    const int tid  = threadIdx.x;
    const int lane = tid & 31;
    const int wave = tid >> 5;
    const int nStrips = N >> 6;
    const int bm = blockIdx.x / nStrips;          // 128-row block tile
    const int bn = blockIdx.x % nStrips;          // 64-col strip
    const int r  = lane & 15;
    const int kh = lane >> 4;

    __shared__ __bf16 ldsB[2][64 * LDSB_STRIDE];  // 2 x 9 KB

    const __bf16* aRow  = A + (size_t)(bm * 128 + wave * 16 + r) * K;
    const __bf16* bBase = B + (size_t)(bn * 64) * K;

    // stage one 64(rows) x 64(k) chunk of B into LDS buffer `buf`
    auto stage = [&](int buf, int kc) {
#pragma unroll
        for (int c = 0; c < 2; ++c) {
            const int idx = tid + c * 256;        // 512 x 16B chunks
            const int row = idx >> 3;             // 0..63
            const int col = (idx & 7) * 8;        // bf16 elems within row
            const __bf16* src = bBase + (size_t)row * K + kc + col;
            __bf16* dst = &ldsB[buf][row * LDSB_STRIDE + col];
            ASYNC_CP16(dst, src);
        }
    };

    stage(0, 0);
    v8f acc[4] = {};
    int buf = 0;
    for (int kc = 0; kc < K; kc += KC) {
        const bool hasNext = (kc + KC) < K;
        if (hasNext) stage(buf ^ 1, kc + KC);
        // async loads complete in order: 2 in flight for the next chunk are
        // allowed while we require the current chunk's 2 to be done.
        if (hasNext) ASYNC_WAIT(2);
        else         ASYNC_WAIT(0);
        __syncthreads();                          // current buffer visible

        const __bf16* bl = ldsB[buf];
#pragma unroll
        for (int k2 = 0; k2 < KC; k2 += 32) {
            v16bf af = load_frag(aRow + kc + k2, kh);
            v16bf bfr[4];
#pragma unroll
            for (int t = 0; t < 4; ++t)
                bfr[t] = load_frag(bl + (t * 16 + r) * LDSB_STRIDE + k2, kh);
#pragma unroll
            for (int t = 0; t < 4; ++t)
                acc[t] = wmma_bf16(af, bfr[t], acc[t]);
        }
        __syncthreads();                          // done reading before rewrite
        buf ^= 1;
    }

    // C/D layout: row = v + 8*kh, col = lane&15
#pragma unroll
    for (int t = 0; t < 4; ++t) {
#pragma unroll
        for (int v = 0; v < 8; ++v) {
            const int m = bm * 128 + wave * 16 + v + 8 * kh;
            const int n = bn * 64 + t * 16 + r;
            C[(size_t)m * N + n] = acc[t][v];
        }
    }
}

// ---- RoPE (pair-adjacent / llama complex layout) + cast to bf16 ----------
__global__ void rope_cast_kernel(const float* __restrict__ q32,
                                 const float* __restrict__ k32,
                                 const float* __restrict__ fc,   // (s, 64, 2)
                                 __bf16* __restrict__ qb,
                                 __bf16* __restrict__ kb, int total) {
    int i = blockIdx.x * blockDim.x + threadIdx.x;
    if (i >= total) return;                 // total = ROWS * DMODEL/2
    const int pair = i % (DMODEL / 2);
    const int row  = i / (DMODEL / 2);
    const int si   = row % SEQ;
    const int dh   = pair & (HEADDIM / 2 - 1);     // rotation index in head
    const float c  = fc[((size_t)si * (HEADDIM / 2) + dh) * 2 + 0];
    const float sn = fc[((size_t)si * (HEADDIM / 2) + dh) * 2 + 1];
    const size_t base = (size_t)row * DMODEL + pair * 2;

    float q1 = q32[base], q2 = q32[base + 1];
    qb[base]     = (__bf16)(q1 * c - q2 * sn);
    qb[base + 1] = (__bf16)(q1 * sn + q2 * c);
    float k1 = k32[base], k2 = k32[base + 1];
    kb[base]     = (__bf16)(k1 * c - k2 * sn);
    kb[base + 1] = (__bf16)(k1 * sn + k2 * c);
}

// ---- V transpose: (b,s,h,d) fp32 -> (b,h,d,s) bf16 -----------------------
__global__ void vtrans_kernel(const float* __restrict__ v32,
                              __bf16* __restrict__ vt, int n) {
    int i = blockIdx.x * blockDim.x + threadIdx.x;
    if (i >= n) return;
    const int d  = i % HEADDIM;
    const int h  = (i / HEADDIM) % NHEADS;
    const int s_ = (i / (HEADDIM * NHEADS)) % SEQ;
    const int b  = i / (HEADDIM * NHEADS * SEQ);
    vt[(((size_t)b * NHEADS + h) * HEADDIM + d) * SEQ + s_] = (__bf16)v32[i];
}

// ---- Block-cooperative flash attention ------------------------------------
// Block = 4 waves = 64 consecutive queries of one (b,h).  K tile (32x128)
// and V tile (128x32) are staged into double-buffered LDS shared by all 4
// waves -- via TDM descriptors (wave 0 issues, TENSORcnt) when available,
// else via per-thread async copies (ASYNCcnt).  kv trip count is
// block-uniform; per-wave compute is predicated by its causal limit.
#define KSTR 136   // K tile row stride (128 + 8 pad) -> distinct banks
#define VSTR 40    // V tile row stride (32 + 8 pad)  -> distinct banks

__global__ __launch_bounds__(128)
void attn_kernel(const __bf16* __restrict__ qb,   // (b,s,h,d)
                 const __bf16* __restrict__ kb,   // (b,s,h,d)
                 const __bf16* __restrict__ vt,   // (b,h,d,s)
                 __bf16* __restrict__ ctx) {      // (b,s, h*hd+d)
    const int tid  = threadIdx.x;
    const int lane = tid & 31;
    const int wave = tid >> 5;
    const int q64  = blockIdx.x & 31;             // SEQ/64 blocks per (b,h)
    const int h    = (blockIdx.x >> 5) & 15;
    const int b    = blockIdx.x >> 9;
    const int qBaseBlk = q64 * 64;
    const int qBase = qBaseBlk + wave * 16;
    const int r  = lane & 15;
    const int kh = lane >> 4;

    __shared__ __bf16 ldsK[2][32 * KSTR];         // 2 x 8.5 KB
    __shared__ __bf16 ldsV[2][HEADDIM * VSTR];    // 2 x 10 KB
    __shared__ __bf16 pLds[4][16 * 32];           // per-wave P staging
    __bf16* myP = pLds[wave];

#if HAVE_TDM
    // one 2D TDM descriptor per K / V tile, issued by wave 0 only.
    auto stageKV = [&](int bufi, int kv) {
        if (wave == 0) {
            // K: 32 rows (keys) x 128 bf16, row pitch DMODEL elems;
            // LDS pad 4 dwords per 64 dwords -> KSTR rows.
            const __bf16* ksrc =
                kb + (((size_t)b * SEQ + kv) * NHEADS + h) * HEADDIM;
            tdm_load_2d(ksrc, lds_addr_of(&ldsK[bufi][0]),
                        HEADDIM, SEQ, DMODEL, HEADDIM, 32, 5u, 3u);
            // V: 128 rows (dims) x 32 bf16 at key offset kv, pitch SEQ;
            // LDS pad 4 dwords per 16 dwords -> VSTR rows.
            const __bf16* vsrc =
                vt + ((size_t)b * NHEADS + h) * HEADDIM * SEQ + kv;
            tdm_load_2d(vsrc, lds_addr_of(&ldsV[bufi][0]),
                        SEQ, HEADDIM, SEQ, 32, HEADDIM, 3u, 3u);
        }
    };
#else
    // per-thread async copies: 8 x b128 per thread per stage
    auto stageKV = [&](int bufi, int kv) {
#pragma unroll
        for (int c = 0; c < 4; ++c) {
            const int lin = tid + c * 128;        // 0..511
            const int krow = lin >> 4;            // 0..31
            const int kseg = (lin & 15) * 8;
            const __bf16* ksrc =
                kb + (((size_t)b * SEQ + kv + krow) * NHEADS + h) * HEADDIM + kseg;
            ASYNC_CP16(&ldsK[bufi][krow * KSTR + kseg], ksrc);

            const int vrow = lin >> 2;            // 0..127 (head dim)
            const int vseg = (lin & 3) * 8;
            const __bf16* vsrc =
                vt + (((size_t)b * NHEADS + h) * HEADDIM + vrow) * SEQ + kv + vseg;
            ASYNC_CP16(&ldsV[bufi][vrow * VSTR + vseg], vsrc);
        }
    };
#endif

    // cache the 16x128 Q tile as 4 A-frags
    const __bf16* qRow = qb + (((size_t)b * SEQ + qBase + r) * NHEADS + h) * HEADDIM;
    v16bf qf[4];
#pragma unroll
    for (int d0 = 0; d0 < 4; ++d0) qf[d0] = load_frag(qRow + d0 * 32, kh);

    float mrow[8], lrow[8];
    v8f o[8] = {};
#pragma unroll
    for (int v = 0; v < 8; ++v) { mrow[v] = -3.0e38f; lrow[v] = 0.0f; }

    const float scale = 0.08838834764831845f;     // 1/sqrt(128)
    const int kvEndBlk = qBaseBlk + 64;           // block-uniform trip count
    const int kvEndW   = qBase + 16;              // this wave's causal limit

    stageKV(0, 0);
    int buf = 0;
    for (int kv = 0; kv < kvEndBlk; kv += 32) {
        const bool hasNext = (kv + 32) < kvEndBlk;
        if (hasNext) stageKV(buf ^ 1, kv + 32);
#if HAVE_TDM
        // tensor ops are in-order per wave: with 2 DMAs (K+V) in flight for
        // the next tile, tensorcnt<=2 means the current tile is complete.
        if (wave == 0) {
            if (hasNext) __builtin_amdgcn_s_wait_tensorcnt(2);
            else         __builtin_amdgcn_s_wait_tensorcnt(0);
        }
#else
        if (hasNext) ASYNC_WAIT(8);               // 8 in flight for next tile
        else         ASYNC_WAIT(0);
#endif
        __syncthreads();                          // current K/V tiles visible

        if (kv < kvEndW) {
            const __bf16* lk = ldsK[buf];
            const __bf16* lv = ldsV[buf];

            // ---- scores: keys [kv,kv+16) and [kv+16,kv+32)
            v8f s0 = {}, s1 = {};
#pragma unroll
            for (int d0 = 0; d0 < 4; ++d0) {
                v16bf kf0 = load_frag(lk + r * KSTR + d0 * 32, kh);
                v16bf kf1 = load_frag(lk + (16 + r) * KSTR + d0 * 32, kh);
                s0 = wmma_bf16(qf[d0], kf0, s0);
                s1 = wmma_bf16(qf[d0], kf1, s1);
            }

            // ---- online softmax (row = v + 8*kh, col = lane&15)
#pragma unroll
            for (int v = 0; v < 8; ++v) {
                const int qIdx = qBase + v + 8 * kh;
                float x0 = s0[v] * scale;
                float x1 = s1[v] * scale;
                if (kv + r      > qIdx) x0 = -3.0e38f;
                if (kv + 16 + r > qIdx) x1 = -3.0e38f;

                float rm = fmaxf(x0, x1);
#pragma unroll
                for (int off = 1; off < 16; off <<= 1)
                    rm = fmaxf(rm, __shfl_xor(rm, off, 32));

                const float mnew = fmaxf(mrow[v], rm);
                const float corr = __expf(mrow[v] - mnew);
                const float p0 = __expf(x0 - mnew);
                const float p1 = __expf(x1 - mnew);
                float rs = p0 + p1;
#pragma unroll
                for (int off = 1; off < 16; off <<= 1)
                    rs += __shfl_xor(rs, off, 32);

                lrow[v] = lrow[v] * corr + rs;
                mrow[v] = mnew;
#pragma unroll
                for (int t = 0; t < 8; ++t) o[t][v] *= corr;

                const int rowm = v + 8 * kh;
                myP[rowm * 32 + r]      = (__bf16)p0;
                myP[rowm * 32 + 16 + r] = (__bf16)p1;
            }

            // ---- P (16x32) as A-frag (per-wave LDS; DS in-order per wave)
            v16bf pf = load_frag(myP + r * 32, kh);

            // ---- O += P @ Vtile, 8 head-dim tiles from shared LDS V
            v16bf vfr[8];
#pragma unroll
            for (int t = 0; t < 8; ++t)
                vfr[t] = load_frag(lv + (t * 16 + r) * VSTR, kh);
#pragma unroll
            for (int t = 0; t < 8; ++t)
                o[t] = wmma_bf16(pf, vfr[t], o[t]);
        }

        __syncthreads();                          // done reading before rewrite
        buf ^= 1;
    }

    // ---- normalize + store ctx (row-major 4096 x 2048, bf16)
#pragma unroll
    for (int v = 0; v < 8; ++v) {
        const float inv = 1.0f / lrow[v];
        const int qIdx = qBase + v + 8 * kh;
        __bf16* crow = ctx + ((size_t)b * SEQ + qIdx) * DMODEL + h * HEADDIM;
#pragma unroll
        for (int t = 0; t < 8; ++t)
            crow[t * 16 + r] = (__bf16)(o[t][v] * inv);
    }
}

// ---------------------------------------------------------------------------
extern "C" void kernel_launch(void* const* d_in, const int* in_sizes, int n_in,
                              void* d_out, int out_size, void* d_ws, size_t ws_size,
                              hipStream_t stream) {
    const float* x  = (const float*)d_in[0];
    const float* fc = (const float*)d_in[1];
    const float* wq = (const float*)d_in[2];
    const float* wk = (const float*)d_in[3];
    const float* wv = (const float*)d_in[4];
    const float* wo = (const float*)d_in[5];
    float* out = (float*)d_out;

    // ---- workspace carve-up (all sizes are multiples of 256B)
    char* w = (char*)d_ws;
    const size_t nX = (size_t)ROWS * DMODEL;      // 8,388,608
    const size_t nW = (size_t)DMODEL * DMODEL;    // 4,194,304
    __bf16* xb  = (__bf16*)w; w += nX * 2;
    __bf16* wqb = (__bf16*)w; w += nW * 2;
    __bf16* wkb = (__bf16*)w; w += nW * 2;
    __bf16* wvb = (__bf16*)w; w += nW * 2;
    __bf16* wob = (__bf16*)w; w += nW * 2;
    float*  q32 = (float*)w;  w += nX * 4;
    float*  k32 = (float*)w;  w += nX * 4;
    float*  v32 = (float*)w;  w += nX * 4;
    __bf16* qbuf = (__bf16*)w; w += nX * 2;
    __bf16* kbuf = (__bf16*)w; w += nX * 2;
    __bf16* vtb  = (__bf16*)w; w += nX * 2;
    __bf16* ctx  = (__bf16*)w; w += nX * 2;

    // ---- 1. casts to bf16
    cast_bf16_kernel<<<(int)((nX + 255) / 256), 256, 0, stream>>>(x, xb, (int)nX);
    cast_bf16_kernel<<<(int)((nW + 255) / 256), 256, 0, stream>>>(wq, wqb, (int)nW);
    cast_bf16_kernel<<<(int)((nW + 255) / 256), 256, 0, stream>>>(wk, wkb, (int)nW);
    cast_bf16_kernel<<<(int)((nW + 255) / 256), 256, 0, stream>>>(wv, wvb, (int)nW);
    cast_bf16_kernel<<<(int)((nW + 255) / 256), 256, 0, stream>>>(wo, wob, (int)nW);

    // ---- 2. QKV projections: (4096x2048) @ (2048x2048)^T
    const int gemmBlocks = (ROWS / 128) * (DMODEL / 64);      // 32*32 = 1024
    gemm_bf16_nt_kernel<<<gemmBlocks, 256, 0, stream>>>(xb, wqb, q32, ROWS, DMODEL, DMODEL);
    gemm_bf16_nt_kernel<<<gemmBlocks, 256, 0, stream>>>(xb, wkb, k32, ROWS, DMODEL, DMODEL);
    gemm_bf16_nt_kernel<<<gemmBlocks, 256, 0, stream>>>(xb, wvb, v32, ROWS, DMODEL, DMODEL);

    // ---- 3. RoPE on q,k (fp32 math, bf16 out); transpose V to (b,h,d,s)
    const int nPairs = ROWS * DMODEL / 2;
    rope_cast_kernel<<<(nPairs + 255) / 256, 256, 0, stream>>>(q32, k32, fc, qbuf, kbuf, nPairs);
    vtrans_kernel<<<(int)((nX + 255) / 256), 256, 0, stream>>>(v32, vtb, (int)nX);

    // ---- 4. causal flash attention -> ctx bf16 (4 waves share K/V tiles)
    const int attnBlocks = BATCH * NHEADS * (SEQ / 64);       // 1024
    attn_kernel<<<attnBlocks, 128, 0, stream>>>(qbuf, kbuf, vtb, ctx);

    // ---- 5. output projection -> fp32 d_out
    gemm_bf16_nt_kernel<<<gemmBlocks, 256, 0, stream>>>(ctx, wob, out, ROWS, DMODEL, DMODEL);
}
